// ModulatedConv2d_66735201845784
// MI455X (gfx1250) — matrix operand order repaired and checked by reading, not compiled, not run on hardware
//
#include <hip/hip_runtime.h>
#include <hip/hip_bf16.h>

// ---- constants from the reference ----
// B=8, O=64, I=64, K=3, H=128, W=128
// x: (8, 512, 128, 128) f32 ; styles: (8, 64) f32 ; weight: (64, 64, 3, 3) f32
// out: (8, 512, 128, 128) f32

typedef __attribute__((ext_vector_type(16))) __bf16 bf16x16;
typedef __attribute__((ext_vector_type(8)))  __bf16 bf16x8;
typedef __attribute__((ext_vector_type(8)))  float  f32x8;

union V16B {
    bf16x16 v;
    bf16x8  h[2];
};

static __device__ __forceinline__ __bf16 f2bf(float f) {
    union { float f; unsigned int u; } x;
    x.f = f;
    // round-to-nearest-even bf16
    unsigned int r = x.u + 0x7FFFu + ((x.u >> 16) & 1u);
    unsigned short hs = (unsigned short)(r >> 16);
    __bf16 out;
    __builtin_memcpy(&out, &hs, 2);
    return out;
}

// -------------------------------------------------------------------------
// Kernel 1: modulate + demodulate weights -> bf16, layout [g][tap][c][o][i']
//   wbuf index: (((g*9 + t)*2 + c)*64 + o)*32 + ip      (t = kh*3+kw, i = c*32+ip)
//   demod[g][kw] = rsqrt( sum_{o,i,kh} (weight[o,i,kh,kw]*styles[g,i])^2 + 1e-8 )
//   (matches the reference's demod-per-(batch, kw) broadcast quirk)
// -------------------------------------------------------------------------
__global__ void modconv_wprep_kernel(const float* __restrict__ weight,
                                     const float* __restrict__ styles,
                                     __bf16* __restrict__ wbuf) {
    const int g   = blockIdx.x;      // 0..7
    const int tid = threadIdx.x;     // 0..255
    const int NW  = 64 * 64 * 3 * 3; // 36864

    __shared__ float red[3][256];
    __shared__ float dm[3];

    float s0 = 0.f, s1 = 0.f, s2 = 0.f;
    for (int j = tid; j < NW; j += 256) {
        const int kw = j % 3;
        const int i  = (j / 9) & 63;
        const float v = weight[j] * styles[g * 64 + i];
        const float v2 = v * v;
        if (kw == 0) s0 += v2; else if (kw == 1) s1 += v2; else s2 += v2;
    }
    red[0][tid] = s0; red[1][tid] = s1; red[2][tid] = s2;
    __syncthreads();
    for (int stride = 128; stride > 0; stride >>= 1) {
        if (tid < stride) {
            red[0][tid] += red[0][tid + stride];
            red[1][tid] += red[1][tid + stride];
            red[2][tid] += red[2][tid + stride];
        }
        __syncthreads();
    }
    if (tid < 3) dm[tid] = rsqrtf(red[tid][0] + 1e-8f);
    __syncthreads();

    for (int j = tid; j < NW; j += 256) {
        const int o   = j / 576;
        const int rem = j % 576;
        const int i   = rem / 9;
        const int kh  = (rem % 9) / 3;
        const int kw  = rem % 3;
        const float v = weight[j] * styles[g * 64 + i] * dm[kw];
        const int t  = kh * 3 + kw;
        const int c  = i >> 5;
        const int ip = i & 31;
        wbuf[(((g * 9 + t) * 2 + c) * 64 + o) * 32 + ip] = f2bf(v);
    }
}

// -------------------------------------------------------------------------
// Kernel 2: grouped 3x3 conv as implicit GEMM on bf16 WMMA.
// One block = (n, g, h): output tile 64 channels x 128 pixels.
// 256 threads = 8 waves; wave wv owns pixel columns [wv*16, wv*16+16).
// Outer loop over kh (3 input rows). Each input row is staged ONCE into a
// halo-padded LDS tile (columns -1..128); the three kw taps are served by a
// +kw row offset into that tile -> 3x less global traffic than per-tap
// staging. All 3 taps' A tiles for this kh are staged together (24 KB).
// Inner: 3 kw x 2 k-chunks x 4 M-tiles = 24 WMMAs per kh.
// -------------------------------------------------------------------------
__global__ void __launch_bounds__(256, 2)
modconv_gemm_kernel(const float* __restrict__ x,
                    const __bf16* __restrict__ wbuf,
                    float* __restrict__ out) {
    const int h  = blockIdx.x;   // 0..127 output row
    const int g  = blockIdx.y;   // 0..7 group (style batch)
    const int n  = blockIdx.z;   // 0..7 outer batch

    const int tid  = threadIdx.x;
    const int lane = tid & 31;
    const int wv   = tid >> 5;

    // A tiles: weights for 3 taps, [kw][c][o][ip] (rows 64 B, 16B-aligned slices)
    __shared__ __align__(16) __bf16 Alds[3][2][64][32];    // 24 KB
    // B tile: one input row + column halo, [col+1][k], row stride 80 el = 160 B
    __shared__ __align__(16) __bf16 Blds[130][80];         // 20.3 KB

    f32x8 acc[4];
#pragma unroll
    for (int m = 0; m < 4; ++m) acc[m] = (f32x8)0.0f;

    const float* xg = x + (((size_t)n * 512 + (size_t)g * 64) * 128) * 128;

    // lane constants for fragment gathers (per CDNA5 16-bit WMMA layouts)
    const int hiHalf = (lane & 16) ? 1 : 0;
    const int l15    = lane & 15;
    const int col16  = (wv << 4) + l15;        // pixel column this lane produces
    const int aBase  = hiHalf ? 8 : 0;         // A-frag K sub-offset (elements)
    const int bBase  = hiHalf ? 8 : 0;         // B-frag K sub-offset (elements)

    for (int kh = 0; kh < 3; ++kh) {
        const int row = h + kh - 1;
        const bool rowok = (row >= 0) && (row < 128);

        __syncthreads();  // previous iteration's consumers done

        // ---- stage A: 3 taps x 8 KB = 24 KB linear copy, 96 B per thread ----
        {
            const bf16x8* src =
                (const bf16x8*)(wbuf + (size_t)(g * 9 + kh * 3) * 4096);
            bf16x8* dst = (bf16x8*)(&Alds[0][0][0][0]);
#pragma unroll
            for (int j = 0; j < 6; ++j)
                dst[tid * 6 + j] = src[tid * 6 + j];
        }

        // ---- stage B: 64 ch x 130 cols (halo -1..128), zero-padded ----
        {
            const float* xrow = xg + (size_t)(rowok ? row : 0) * 128;
#pragma unroll 4
            for (int e = 0; e < 33; ++e) {
                const int idx = tid + (e << 8);     // 0 .. 8448
                if (idx < 64 * 130) {
                    const int k = idx / 130;        // input channel in group
                    const int c = idx - k * 130;    // LDS column (= input col + 1)
                    const int col = c - 1;
                    float v = 0.0f;
                    if (rowok && col >= 0 && col < 128)
                        v = xrow[(size_t)k * 16384 + col];
                    Blds[c][k] = f2bf(v);
                }
            }
        }

        __syncthreads();

        // ---- 3 kw taps x 2 k-chunks x 4 M-tiles ----
#pragma unroll
        for (int kw = 0; kw < 3; ++kw) {
#pragma unroll
            for (int c2 = 0; c2 < 2; ++c2) {
                // B fragment: column (col16 + kw) of the halo tile,
                // K = c2*32 + bBase .. +7 and +16 .. +23 (16B-aligned)
                const __bf16* bp = &Blds[col16 + kw][c2 * 32 + bBase];
                V16B bb;
                bb.h[0] = *(const bf16x8*)(bp);
                bb.h[1] = *(const bf16x8*)(bp + 16);
                bf16x16 bfrag = bb.v;
#pragma unroll
                for (int m = 0; m < 4; ++m) {
                    const __bf16* ap = &Alds[kw][c2][m * 16 + l15][0];
                    V16B a;
                    a.h[0] = *(const bf16x8*)(ap + aBase);       // K base..+7
                    a.h[1] = *(const bf16x8*)(ap + aBase + 16);  // K base+16..+23
                    acc[m] = __builtin_amdgcn_wmma_f32_16x16x32_bf16(
                        /*neg_a=*/false, a.v,
                        /*neg_b=*/false, bfrag,
                        /*c_mod=*/(short)0, acc[m],
                        /*reuse_a=*/false, /*reuse_b=*/false);
                }
            }
        }
    }

    // ---- write out: C/D layout: VGPR r -> M = r + 8*(lane/16), N = lane%16 ----
    const size_t obase = (((size_t)n * 512 + (size_t)g * 64) * 128 + h) * 128;
#pragma unroll
    for (int m = 0; m < 4; ++m) {
#pragma unroll
        for (int r = 0; r < 8; ++r) {
            const int o = m * 16 + r + (hiHalf << 3);
            out[obase + (size_t)o * 16384 + col16] = acc[m][r];
        }
    }
}

// -------------------------------------------------------------------------
extern "C" void kernel_launch(void* const* d_in, const int* in_sizes, int n_in,
                              void* d_out, int out_size, void* d_ws, size_t ws_size,
                              hipStream_t stream) {
    const float* x      = (const float*)d_in[0];   // (8, 512, 128, 128)
    const float* styles = (const float*)d_in[1];   // (8, 64)
    const float* weight = (const float*)d_in[2];   // (64, 64, 3, 3)
    float* out          = (float*)d_out;           // (8, 512, 128, 128)

    __bf16* wbuf = (__bf16*)d_ws;                  // 8*9*2*64*32 bf16 = 576 KB

    modconv_wprep_kernel<<<dim3(8), dim3(256), 0, stream>>>(weight, styles, wbuf);
    modconv_gemm_kernel<<<dim3(128, 8, 8), dim3(256), 0, stream>>>(x, wbuf, out);
}
